// ProTCL_8486855377461
// MI455X (gfx1250) — matrix operand order repaired and checked by reading, not compiled, not run on hardware
//
#include <hip/hip_runtime.h>
#include <hip/hip_bf16.h>

typedef __bf16 v16bf __attribute__((ext_vector_type(16)));
typedef float  v8f   __attribute__((ext_vector_type(8)));

union Frag16 {
  v16bf v;
  uint4 q[2];
};

__device__ __forceinline__ unsigned short f32_to_bf16_rne(float f) {
  unsigned int u = __float_as_uint(f);
  u += 0x7FFFu + ((u >> 16) & 1u);
  return (unsigned short)(u >> 16);
}

__device__ __forceinline__ unsigned int pack_bf16x2(float lo, float hi) {
  return (unsigned int)f32_to_bf16_rne(lo) |
         ((unsigned int)f32_to_bf16_rne(hi) << 16);
}

// ---------------------------------------------------------------------------
// Convert a [rows x cols] slice of a row-major fp32 matrix (leading dim ld,
// column offset off) into a packed row-major bf16 matrix.
// ---------------------------------------------------------------------------
__global__ void cvt_bf16_kernel(const float* __restrict__ src,
                                unsigned short* __restrict__ dst,
                                int rows, int cols, int ld, int off) {
  int i = blockIdx.x * blockDim.x + threadIdx.x;
  if (i >= rows * cols) return;
  int r = i / cols, c = i - r * cols;
  dst[i] = f32_to_bf16_rne(src[(size_t)r * ld + off + c]);
}

// ---------------------------------------------------------------------------
// Tiny fp32 GEMM  C[M,N] = A[M,K(lda)] * B[N,K(ldb,+off)]^T  (P-path only).
// ---------------------------------------------------------------------------
__global__ void dot_kernel(const float* __restrict__ A, int lda,
                           const float* __restrict__ Bm, int ldb, int off,
                           float* __restrict__ C, int M, int N, int K) {
  int i = blockIdx.x * blockDim.x + threadIdx.x;
  if (i >= M * N) return;
  int m = i / N, n = i - m * N;
  const float* a = A + (size_t)m * lda;
  const float* b = Bm + (size_t)n * ldb + off;
  float s = 0.0f;
  for (int k = 0; k < K; ++k) s = fmaf(a[k], b[k], s);
  C[i] = s;
}

// ---------------------------------------------------------------------------
// WMMA GEMM: C[M,N] = A[M,K] (fp32, converted on the fly) * Bbf[N,K]^T (bf16).
// Block = 256 threads (8 waves). Tile = 64 rows x 128 cols; each wave owns one
// 16-wide N-tile, iterating 4 M-tiles. A staged through LDS (stride 40 bf16,
// bank-conflict-free for the A-fragment ds_load_b128 pattern).
// ---------------------------------------------------------------------------
__global__ __launch_bounds__(256) void gemm_wmma_bf16(
    const float* __restrict__ A, int M, int K, int lda,
    const unsigned short* __restrict__ Bbf,
    float* __restrict__ C, int ldc) {
  __shared__ unsigned short sA[64 * 40];  // 5 KB, stride 40 bf16

  const int tid  = threadIdx.x;
  const int lane = tid & 31;
  const int wave = tid >> 5;
  const int half = lane >> 4;   // 0: lanes 0-15, 1: lanes 16-31
  const int l16  = lane & 15;
  const int m0   = blockIdx.x * 64;
  const int ncol = (blockIdx.y * 8 + wave) * 16 + l16;  // output column

  v8f acc[4];
#pragma unroll
  for (int mt = 0; mt < 4; ++mt) acc[mt] = (v8f)0.0f;

  const int nkc = K >> 5;
  for (int kc = 0; kc < nkc; ++kc) {
    __syncthreads();
    {  // stage 64x32 fp32 -> bf16 into LDS (8 elements per thread)
      int r  = tid & 63;
      int cb = (tid >> 6) * 8;
      int grow = m0 + r;
      unsigned short* dst = &sA[r * 40 + cb];
      if (grow < M) {
        const float* src = A + (size_t)grow * lda + kc * 32 + cb;
#pragma unroll
        for (int i = 0; i < 8; ++i) dst[i] = f32_to_bf16_rne(src[i]);
      } else {
#pragma unroll
        for (int i = 0; i < 8; ++i) dst[i] = 0;
      }
    }
    __syncthreads();

    // B fragment: lane holds col ncol, K = half*8..half*8+7, 16+half*8..16+half*8+7
    Frag16 bf;
    const uint4* bp =
        (const uint4*)(Bbf + (size_t)ncol * K + kc * 32 + half * 8);
    bf.q[0] = bp[0];
    bf.q[1] = bp[2];

#pragma unroll
    for (int mt = 0; mt < 4; ++mt) {
      Frag16 af;
      const uint4* ap = (const uint4*)&sA[(mt * 16 + l16) * 40 + half * 8];
      af.q[0] = ap[0];
      af.q[1] = ap[2];  // +16 bf16 elements = +32 bytes
      acc[mt] = __builtin_amdgcn_wmma_f32_16x16x32_bf16(
          false, af.v, false, bf.v, (short)0, acc[mt], false, false);
    }
  }

  // C layout: VGPR j -> row j + 8*half, col = l16 within the 16x16 tile
#pragma unroll
  for (int mt = 0; mt < 4; ++mt)
#pragma unroll
    for (int j = 0; j < 8; ++j) {
      int row = m0 + mt * 16 + j + half * 8;
      if (row < M) C[(size_t)row * ldc + ncol] = acc[mt][j];
    }
}

// ---------------------------------------------------------------------------
// Fused main kernel: for 64 (b,l) rows per workgroup
//   a1 = relu(hp[b] + hl[l] + b1)            -> bf16 A-tile in LDS (132 KB)
//   h2 = relu(a1 @ W2^T + b2)                -> WMMA accumulators
//   logit = h2 . W3 + b3                     -> shfl + LDS-atomic reduction
// Each wave processes TWO 16-wide N-tiles concurrently so every A-fragment
// LDS load feeds two WMMAs (8 WMMAs per 8 ds_load_b128 per K-step).
// ---------------------------------------------------------------------------
__global__ __launch_bounds__(256) void fused_kernel(
    const float* __restrict__ hp,   // [16,1024]
    const float* __restrict__ hl,   // [5000,1024]
    const float* __restrict__ b1v,  // [1024]
    const unsigned short* __restrict__ W2bf,  // [1024,1024] bf16
    const float* __restrict__ b2v,  // [1024]
    const float* __restrict__ W3v,  // [1024]
    const float* __restrict__ b3v,  // [1]
    float* __restrict__ out) {      // [80000]
  extern __shared__ char smem[];
  float* slog = (float*)smem;                          // 64 floats
  unsigned short* sA = (unsigned short*)(smem + 256);  // 64 x 1032 bf16

  const int tid  = threadIdx.x;
  const int lane = tid & 31;
  const int wave = tid >> 5;
  const int half = lane >> 4;
  const int l16  = lane & 15;
  const int r0   = blockIdx.x * 64;

  if (tid < 64) slog[tid] = 0.0f;

  {  // Phase 1: build bf16 A tile. thread -> row tid&63, 256-col quarter.
    int m  = tid & 63;
    int cb = (tid >> 6) * 256;
    int r  = r0 + m;
    int b  = r / 5000;
    int l  = r - b * 5000;
    const float4* hpp = (const float4*)(hp + (size_t)b * 1024 + cb);
    const float4* hlp = (const float4*)(hl + (size_t)l * 1024 + cb);
    const float4* b1p = (const float4*)(b1v + cb);
    unsigned int* dst = (unsigned int*)&sA[m * 1032 + cb];
#pragma unroll 4
    for (int i = 0; i < 64; ++i) {
      float4 a = hpp[i], c = hlp[i], d = b1p[i];
      float x0 = fmaxf(a.x + c.x + d.x, 0.0f);
      float x1 = fmaxf(a.y + c.y + d.y, 0.0f);
      float x2 = fmaxf(a.z + c.z + d.z, 0.0f);
      float x3 = fmaxf(a.w + c.w + d.w, 0.0f);
      dst[2 * i]     = pack_bf16x2(x0, x1);
      dst[2 * i + 1] = pack_bf16x2(x2, x3);
    }
  }
  __syncthreads();

  // Phase 2: 4 pairs of N-tiles per wave; K = 1024 in 32-element steps.
  float part[4][8];
#pragma unroll
  for (int mt = 0; mt < 4; ++mt)
#pragma unroll
    for (int j = 0; j < 8; ++j) part[mt][j] = 0.0f;

  for (int it = 0; it < 4; ++it) {
    const int ncolA = (it * 16 + wave) * 16 + l16;        // N-tile pair
    const int ncolB = ncolA + 128;                        // (+8 tiles)
    const float b2A = b2v[ncolA], w3A = W3v[ncolA];
    const float b2B = b2v[ncolB], w3B = W3v[ncolB];
    const unsigned short* baseA = W2bf + (size_t)ncolA * 1024 + half * 8;
    const unsigned short* baseB = W2bf + (size_t)ncolB * 1024 + half * 8;

    v8f accA[4], accB[4];
#pragma unroll
    for (int mt = 0; mt < 4; ++mt) {
      accA[mt] = (v8f)0.0f;
      accB[mt] = (v8f)0.0f;
    }

    for (int kc = 0; kc < 32; ++kc) {
      Frag16 bfA, bfB;
      {
        const uint4* bpA = (const uint4*)(baseA + kc * 32);
        const uint4* bpB = (const uint4*)(baseB + kc * 32);
        bfA.q[0] = bpA[0];
        bfA.q[1] = bpA[2];
        bfB.q[0] = bpB[0];
        bfB.q[1] = bpB[2];
      }
#pragma unroll
      for (int mt = 0; mt < 4; ++mt) {
        Frag16 af;
        const uint4* ap =
            (const uint4*)&sA[(mt * 16 + l16) * 1032 + kc * 32 + half * 8];
        af.q[0] = ap[0];
        af.q[1] = ap[2];
        accA[mt] = __builtin_amdgcn_wmma_f32_16x16x32_bf16(
            false, af.v, false, bfA.v, (short)0, accA[mt], false, false);
        accB[mt] = __builtin_amdgcn_wmma_f32_16x16x32_bf16(
            false, af.v, false, bfB.v, (short)0, accB[mt], false, false);
      }
    }
    // epilogue for this N-tile pair: relu(h2 + b2) * W3, accumulated per row
#pragma unroll
    for (int mt = 0; mt < 4; ++mt)
#pragma unroll
      for (int j = 0; j < 8; ++j)
        part[mt][j] += fmaxf(accA[mt][j] + b2A, 0.0f) * w3A +
                       fmaxf(accB[mt][j] + b2B, 0.0f) * w3B;
  }

  // Reduce the 16 lanes of each half-wave (they hold 16 different columns of
  // the same row), then accumulate across waves via LDS atomics.
#pragma unroll
  for (int mt = 0; mt < 4; ++mt)
#pragma unroll
    for (int j = 0; j < 8; ++j) {
      float p = part[mt][j];
      p += __shfl_xor(p, 1, 32);
      p += __shfl_xor(p, 2, 32);
      p += __shfl_xor(p, 4, 32);
      p += __shfl_xor(p, 8, 32);
      if (l16 == 0) atomicAdd(&slog[mt * 16 + j + half * 8], p);
    }
  __syncthreads();

  if (tid < 64) out[r0 + tid] = slog[tid] + b3v[0];
}

// ---------------------------------------------------------------------------
extern "C" void kernel_launch(void* const* d_in, const int* in_sizes, int n_in,
                              void* d_out, int out_size, void* d_ws,
                              size_t ws_size, hipStream_t stream) {
  (void)in_sizes; (void)n_in; (void)out_size; (void)ws_size;
  const float* seq = (const float*)d_in[0];  // [16,1100]
  const float* lab = (const float*)d_in[1];  // [5000,1024]
  const float* Wp  = (const float*)d_in[2];  // [1024,1100]
  const float* Wl  = (const float*)d_in[3];  // [1024,1024]
  const float* W1  = (const float*)d_in[4];  // [1024,2048]
  const float* b1  = (const float*)d_in[5];  // [1024]
  const float* W2  = (const float*)d_in[6];  // [1024,1024]
  const float* b2  = (const float*)d_in[7];  // [1024]
  const float* W3  = (const float*)d_in[8];  // [1,1024]
  const float* b3  = (const float*)d_in[9];  // [1]
  float* out = (float*)d_out;

  char* ws = (char*)d_ws;
  float* Pe = (float*)ws;                 ws += (size_t)16 * 1024 * 4;
  float* hp = (float*)ws;                 ws += (size_t)16 * 1024 * 4;
  float* Le = (float*)ws;                 ws += (size_t)5000 * 1024 * 4;
  float* hl = (float*)ws;                 ws += (size_t)5000 * 1024 * 4;
  unsigned short* Wlbf  = (unsigned short*)ws; ws += (size_t)1024 * 1024 * 2;
  unsigned short* W1lbf = (unsigned short*)ws; ws += (size_t)1024 * 1024 * 2;
  unsigned short* W2bf  = (unsigned short*)ws; ws += (size_t)1024 * 1024 * 2;

  // Weight conversions (fp32 -> bf16)
  const int cvtBlocks = (1024 * 1024 + 255) / 256;
  cvt_bf16_kernel<<<cvtBlocks, 256, 0, stream>>>(Wl, Wlbf, 1024, 1024, 1024, 0);
  cvt_bf16_kernel<<<cvtBlocks, 256, 0, stream>>>(W1, W1lbf, 1024, 1024, 2048, 1024);
  cvt_bf16_kernel<<<cvtBlocks, 256, 0, stream>>>(W2, W2bf, 1024, 1024, 1024, 0);

  // P-path (tiny, fp32): Pe = seq @ Wp^T ; hp = Pe @ W1[:, :1024]^T
  dot_kernel<<<64, 256, 0, stream>>>(seq, 1100, Wp, 1100, 0, Pe, 16, 1024, 1100);
  dot_kernel<<<64, 256, 0, stream>>>(Pe, 1024, W1, 2048, 0, hp, 16, 1024, 1024);

  // L-path (WMMA): Le = lab @ Wl^T ; hl = Le @ W1[:, 1024:]^T
  dim3 gL((5000 + 63) / 64, 1024 / 128);
  gemm_wmma_bf16<<<gL, 256, 0, stream>>>(lab, 5000, 1024, 1024, Wlbf, Le, 1024);
  gemm_wmma_bf16<<<gL, 256, 0, stream>>>(Le, 5000, 1024, 1024, W1lbf, hl, 1024);

  // Fused cross-join MLP: 80000 rows / 64 per workgroup
  const size_t smemBytes = 256 + (size_t)64 * 1032 * 2;  // 132352 B
  hipFuncSetAttribute((const void*)fused_kernel,
                      hipFuncAttributeMaxDynamicSharedMemorySize,
                      (int)smemBytes);
  fused_kernel<<<1250, 256, smemBytes, stream>>>(hp, hl, b1, W2bf, b2, W3, b3,
                                                 out);
}
